// SpikeConv2d_8469675508098
// MI455X (gfx1250) — matrix operand order repaired and checked by reading, not compiled, not run on hardware
//
#include <hip/hip_runtime.h>

typedef __attribute__((ext_vector_type(16))) __bf16          v16bf;
typedef __attribute__((ext_vector_type(16))) unsigned short  u16x16;
typedef __attribute__((ext_vector_type(8)))  unsigned short  u16x8;
typedef __attribute__((ext_vector_type(8)))  float           v8f;

#define H_      56
#define W_      56
#define CIN     128
#define CIN_P   136                        // padded LDS stride (272B, breaks bank rotation)
#define COUT    256
#define TSTEPS  8
#define BATCH   8
#define TILE_H  2
#define TILE_W  8
#define PATCH_H 4
#define PATCH_W 10
#define PATCH_POS (PATCH_H * PATCH_W)      // 40 positions (incl. halo)
#define SPATIAL (H_ * W_)                  // 3136

__device__ __forceinline__ unsigned short f2bf(float f) {
    union { float f; unsigned u; } v; v.f = f;
    unsigned r = v.u + 0x7FFFu + ((v.u >> 16) & 1u);   // round-to-nearest-even
    return (unsigned short)(r >> 16);
}

// -------- weight prep: OIHW fp32 -> [tap][co][ci] bf16 (into d_ws) --------
__global__ void wprep_kernel(const float* __restrict__ w,
                             unsigned short* __restrict__ wbf) {
    int idx = blockIdx.x * blockDim.x + threadIdx.x;
    const int total = 9 * COUT * CIN;
    if (idx >= total) return;
    int tap = idx / (COUT * CIN);
    int rem = idx % (COUT * CIN);
    int co  = rem / CIN;
    int ci  = rem % CIN;
    int kh  = tap / 3, kw = tap % 3;
    wbf[idx] = f2bf(w[(((size_t)co * CIN + ci) * 3 + kh) * 3 + kw]);
}

// -------- fused implicit-GEMM conv (bf16 WMMA) + integrate-fire scan --------
// Loop nest: stage all 8 timesteps to LDS once, then tap -> kc -> t so the
// weight (B) fragments are register-resident and reused by 8 WMMAs each.
__global__ __launch_bounds__(256, 1)
void spike_conv_kernel(const float* __restrict__ x,
                       const unsigned short* __restrict__ wbf,
                       const float* __restrict__ bias,
                       float* __restrict__ out) {
    __shared__ unsigned short smem[TSTEPS * PATCH_POS * CIN_P];  // ~85 KB

    const int tid    = threadIdx.x;
    const int wv     = tid >> 5;        // wave id 0..7 -> 32-channel chunk
    const int l      = tid & 31;        // lane
    const int lane_n = l & 15;          // N (channel-within-16) per WMMA layout
    const int hsel   = l >> 4;          // lane half: K sub-block selector

    const int bid = blockIdx.x;         // 8 * 196 blocks
    const int b   = bid / 196;
    const int s   = bid % 196;
    const int th  = s / 7, tw = s % 7;
    const int h0  = th * TILE_H, w0 = tw * TILE_W;

    const int co_base = wv * 32;
    const float bias0 = bias[co_base + lane_n];
    const float bias1 = bias[co_base + 16 + lane_n];

    // A-fragment spatial position of this lane (M index = lane_n)
    const int mr = lane_n >> 3, mc = lane_n & 7;

    // ---- stage zero-padded input halo patches for ALL timesteps into LDS ----
    for (int t = 0; t < TSTEPS; ++t) {
        const float* xb = x + ((size_t)(b * TSTEPS + t) * CIN) * SPATIAL;
        unsigned short* sm = smem + t * (PATCH_POS * CIN_P);
        for (int idx = tid; idx < PATCH_POS * CIN; idx += 256) {
            int ci  = idx / PATCH_POS;
            int pos = idx % PATCH_POS;
            int rr = pos / PATCH_W, cc = pos % PATCH_W;
            int hh = h0 - 1 + rr,   ww = w0 - 1 + cc;
            float v = 0.f;
            if (hh >= 0 && hh < H_ && ww >= 0 && ww < W_)
                v = xb[(size_t)ci * SPATIAL + hh * W_ + ww];
            sm[pos * CIN_P + ci] = f2bf(v);
        }
    }
    __syncthreads();

    // ---- accumulators: one pair (2 x 16 channels) per timestep ----
    v8f acc0[TSTEPS], acc1[TSTEPS];
#pragma unroll
    for (int t = 0; t < TSTEPS; ++t) {
        acc0[t] = (v8f){0.f, 0.f, 0.f, 0.f, 0.f, 0.f, 0.f, 0.f};
        acc1[t] = acc0[t];
    }

    for (int tap = 0; tap < 9; ++tap) {
        const int dh = tap / 3, dw = tap % 3;
        const int abase = ((mr + dh) * PATCH_W + (mc + dw)) * CIN_P;
        const unsigned short* wt = wbf + (size_t)tap * (COUT * CIN);
        if (tap < 8)   // pull next tap's weights toward the WGP
            __builtin_prefetch(wbf + (size_t)(tap + 1) * (COUT * CIN)
                                   + (co_base + lane_n) * CIN, 0, 1);
#pragma unroll
        for (int kc = 0; kc < 4; ++kc) {
            const int k0 = kc * 32;
            // ---- B fragments (32x16 bf16): loaded ONCE, reused by 8 WMMAs ----
            v16bf bf0 = *(const v16bf*)(wt + (co_base + lane_n) * CIN
                                           + k0 + hsel * 16);
            v16bf bf1 = *(const v16bf*)(wt + (co_base + 16 + lane_n) * CIN
                                           + k0 + hsel * 16);
#pragma unroll
            for (int t = 0; t < TSTEPS; ++t) {
                const unsigned short* sm = smem + t * (PATCH_POS * CIN_P) + abase;
                // A fragment (16x32 bf16) from LDS in contiguous k order (k = 16*hsel + s), same as B
                u16x8 a0 = *(const u16x8*)(sm + k0 + hsel * 16);      // K 0-7 / 16-23
                u16x8 a1 = *(const u16x8*)(sm + k0 + hsel * 16 + 8);  // K 8-15 / 24-31
                u16x16 au = __builtin_shufflevector(a0, a1,
                    0, 1, 2, 3, 4, 5, 6, 7, 8, 9, 10, 11, 12, 13, 14, 15);
                v16bf af = __builtin_bit_cast(v16bf, au);
                acc0[t] = __builtin_amdgcn_wmma_f32_16x16x32_bf16(
                    false, af, false, bf0, (short)0, acc0[t], false, false);
                acc1[t] = __builtin_amdgcn_wmma_f32_16x16x32_bf16(
                    false, af, false, bf1, (short)0, acc1[t], false, false);
            }
        }
    }

    // ---- integrate (memb += o + bias), fire (>1), reset, store spikes ----
    v8f memb0 = {0.f, 0.f, 0.f, 0.f, 0.f, 0.f, 0.f, 0.f};
    v8f memb1 = memb0;
#pragma unroll
    for (int t = 0; t < TSTEPS; ++t) {
        float* ob0 = out + ((size_t)((t * BATCH + b) * COUT + co_base + lane_n)) * SPATIAL;
        float* ob1 = ob0 + (size_t)16 * SPATIAL;
#pragma unroll
        for (int j = 0; j < 8; ++j) {
            int mm  = j + hsel * 8;                     // M index per C/D layout
            int off = (h0 + (mm >> 3)) * W_ + (w0 + (mm & 7));
            float mb = memb0[j] + acc0[t][j] + bias0;
            ob0[off] = (mb > 1.0f) ? 1.0f : 0.0f;
            memb0[j] = (mb > 1.0f) ? 0.0f : mb;
            mb = memb1[j] + acc1[t][j] + bias1;
            ob1[off] = (mb > 1.0f) ? 1.0f : 0.0f;
            memb1[j] = (mb > 1.0f) ? 0.0f : mb;
        }
    }
}

extern "C" void kernel_launch(void* const* d_in, const int* in_sizes, int n_in,
                              void* d_out, int out_size, void* d_ws, size_t ws_size,
                              hipStream_t stream) {
    const float* x      = (const float*)d_in[0];   // (64,128,56,56) fp32
    const float* weight = (const float*)d_in[1];   // (256,128,3,3) fp32
    const float* bias_p = (const float*)d_in[2];   // (256,) fp32
    (void)in_sizes; (void)n_in; (void)ws_size;

    unsigned short* wbf = (unsigned short*)d_ws;   // 9*256*128 bf16 = 576 KB
    float* out = (float*)d_out;                    // (64,256,56,56) fp32

    const int wtotal = 9 * COUT * CIN;
    wprep_kernel<<<(wtotal + 255) / 256, 256, 0, stream>>>(weight, wbf);

    const int nblocks = BATCH * (H_ / TILE_H) * (W_ / TILE_W);  // 8*28*7 = 1568
    spike_conv_kernel<<<nblocks, 256, 0, stream>>>(x, wbf, bias_p, out);
}
